// BaseSingleSplitDNAMiteModel_47493748359457
// MI455X (gfx1250) — matrix operand order repaired and checked by reading, not compiled, not run on hardware
//
#include <hip/hip_runtime.h>

#define NFEAT   32
#define NEMB    32
#define NHID    64
#define NPAIRS  496
#define BATCH   2048
#define NTILES  (BATCH / 16)       // 128

typedef __attribute__((ext_vector_type(16))) __bf16 bf16x16;
typedef __attribute__((ext_vector_type(8)))  float  f32x8;

union Frag {
    bf16x16 v;
    __bf16  h[16];
    uint4   q[2];
};

// Build a 16x32 bf16 A-fragment slice for this lane from one 32-float row.
// src already points at row + kbA (kbA = 0 for lanes 0-15, 8 for lanes 16-31).
// ISA A layout: elements 0..7 = K kb..kb+7, elements 8..15 = K kb+16..kb+23.
__device__ __forceinline__ void loadA(Frag& f, const float* src) {
    float4 a = *(const float4*)(src + 0);
    float4 b = *(const float4*)(src + 4);
    float4 c = *(const float4*)(src + 16);
    float4 d = *(const float4*)(src + 20);
    f.h[0]  = (__bf16)a.x; f.h[1]  = (__bf16)a.y; f.h[2]  = (__bf16)a.z; f.h[3]  = (__bf16)a.w;
    f.h[4]  = (__bf16)b.x; f.h[5]  = (__bf16)b.y; f.h[6]  = (__bf16)b.z; f.h[7]  = (__bf16)b.w;
    f.h[8]  = (__bf16)c.x; f.h[9]  = (__bf16)c.y; f.h[10] = (__bf16)c.z; f.h[11] = (__bf16)c.w;
    f.h[12] = (__bf16)d.x; f.h[13] = (__bf16)d.y; f.h[14] = (__bf16)d.z; f.h[15] = (__bf16)d.w;
}

__global__ void zero_f32(float* p, int n) {
    int i = blockIdx.x * blockDim.x + threadIdx.x;
    if (i < n) p[i] = 0.0f;
}

// KT0 = number of K=32 steps in layer 0 (1 for main effects, 2 for pairs).
template <int KT0>
__global__ __launch_bounds__(256) void dnamite_wmma(
    const int* __restrict__ idxTbl,      // mains [B,F]  or pairs [B,P,2]
    const int* __restrict__ pairs_list, const int* __restrict__ offsets,
    const float* __restrict__ embedding,
    const float* __restrict__ gW0, const float* __restrict__ gW1, const float* __restrict__ gW2,
    const float* __restrict__ gB0, const float* __restrict__ gB1, const float* __restrict__ gB2,
    const float* __restrict__ gZ,
    float* __restrict__ out)
{
    constexpr int  K0     = KT0 * 32;
    constexpr bool isMain = (KT0 == 1);

    __shared__ __align__(16) __bf16 sW0[NHID * K0];     // [n][k] transposed
    __shared__ __align__(16) __bf16 sW1[NHID * NHID];   // [n][k] transposed
    __shared__ __align__(16) __bf16 sH1[8][16 * NHID];  // per-wave H1 staging
    __shared__ float sB0[NHID], sB1[NHID], sW2[NHID];
    __shared__ float sB2, sGate;
    __shared__ int   sOff0, sOff1;

    const int unit = blockIdx.x;
    const int tid  = threadIdx.x;

    const float* W0 = gW0 + (size_t)unit * K0 * NHID;
    const float* W1 = gW1 + (size_t)unit * NHID * NHID;
    const float* W2 = gW2 + (size_t)unit * NHID;
    const float* B0 = gB0 + (size_t)unit * NHID;
    const float* B1 = gB1 + (size_t)unit * NHID;

    // ---- stage weights (transposed [n][k]) into LDS as bf16 ----
    for (int e = tid; e < NHID * K0; e += 256) {
        int n = e / K0, k = e - n * K0;
        sW0[n * K0 + k] = (__bf16)W0[k * NHID + n];
    }
    for (int e = tid; e < NHID * NHID; e += 256) {
        int n = e >> 6, k = e & 63;
        sW1[(n << 6) + k] = (__bf16)W1[(k << 6) + n];
    }
    if (tid < NHID) { sB0[tid] = B0[tid]; sB1[tid] = B1[tid]; sW2[tid] = W2[tid]; }
    if (tid == 0) {
        sB2 = gB2[unit];
        float z = gZ[unit];
        float g;
        if      (z <= -0.5f) g = 0.0f;
        else if (z >=  0.5f) g = 1.0f;
        else                 g = -2.0f * z * z * z + 1.5f * z + 0.5f;   // gamma = 1
        sGate = g;
        if (isMain) { sOff0 = offsets[unit]; sOff1 = 0; }
        else {
            int fi = pairs_list[2 * unit], fj = pairs_list[2 * unit + 1];
            sOff0 = offsets[fi]; sOff1 = offsets[fj];
        }
    }
    __syncthreads();

    const int wave = tid >> 5;
    const int lane = tid & 31;
    const int lrow = lane & 15;        // M for A, N for B / output
    const int hi   = lane >> 4;        // lane half
    const int kbA  = hi ? 8 : 0;       // A-fragment K base
    const int kbB  = hi ? 8 : 0;       // B-fragment K base (table order, same as A)
    __bf16* stage = sH1[wave];
    const float gate = sGate, b2 = sB2;
    const int off0 = sOff0, off1 = sOff1;

    // ---- hoist per-unit weight fragments + bias/w2 into registers ----
    Frag bw0[KT0][4];
    #pragma unroll
    for (int t = 0; t < KT0; ++t)
        #pragma unroll
        for (int nt = 0; nt < 4; ++nt) {
            const __bf16* s = &sW0[(nt * 16 + lrow) * K0 + t * 32 + kbB];
            bw0[t][nt].q[0] = *(const uint4*)(s);
            bw0[t][nt].q[1] = *(const uint4*)(s + 16);
        }
    Frag bw1[2][4];
    #pragma unroll
    for (int t = 0; t < 2; ++t)
        #pragma unroll
        for (int nt = 0; nt < 4; ++nt) {
            const __bf16* s = &sW1[(nt * 16 + lrow) * NHID + t * 32 + kbB];
            bw1[t][nt].q[0] = *(const uint4*)(s);
            bw1[t][nt].q[1] = *(const uint4*)(s + 16);
        }
    float bias0[4], bias1[4], w2v[4];
    #pragma unroll
    for (int nt = 0; nt < 4; ++nt) {
        bias0[nt] = sB0[nt * 16 + lrow];
        bias1[nt] = sB1[nt * 16 + lrow];
        w2v[nt]   = sW2[nt * 16 + lrow];
    }

    const int tilesPerBlock = NTILES / gridDim.y;
    const int tileBeg = blockIdx.y * tilesPerBlock;
    const int tileEnd = tileBeg + tilesPerBlock;

    for (int tile = tileBeg + wave; tile < tileEnd; tile += 8) {
        const int b = tile * 16 + lrow;

        // ---- gather embedding rows -> A0 fragments ----
        Frag a0[KT0];
        if constexpr (isMain) {
            int row = idxTbl[b * NFEAT + unit] + off0;
            loadA(a0[0], embedding + (size_t)row * NEMB + kbA);
        } else {
            const int base = (b * NPAIRS + unit) * 2;
            if (tile + 8 < tileEnd) {   // prefetch next tile's indices
                const int bn = (tile + 8) * 16 + lrow;
                __builtin_prefetch(&idxTbl[(bn * NPAIRS + unit) * 2], 0, 1);
            }
            int r0 = idxTbl[base + 0] + off0;
            int r1 = idxTbl[base + 1] + off1;
            loadA(a0[0], embedding + (size_t)r0 * NEMB + kbA);
            loadA(a0[1], embedding + (size_t)r1 * NEMB + kbA);
        }

        // ---- layer 0: X[16,K0] x W0[K0,64] ----
        #pragma unroll
        for (int nt = 0; nt < 4; ++nt) {
            f32x8 acc = {};
            #pragma unroll
            for (int t = 0; t < KT0; ++t)
                acc = __builtin_amdgcn_wmma_f32_16x16x32_bf16(false, a0[t].v, false, bw0[t][nt].v,
                                                              (short)0, acc, false, false);
            #pragma unroll
            for (int i = 0; i < 8; ++i) {
                float hv = fmaxf(acc[i] + bias0[nt], 0.0f);
                stage[(i + hi * 8) * NHID + nt * 16 + lrow] = (__bf16)hv;   // C-layout -> row-major
            }
        }

        asm volatile("s_wait_dscnt 0" ::: "memory");        // staging visible wave-wide

        // ---- H1 staging -> A1 fragments (lane transpose via LDS) ----
        Frag a1[2];
        {
            const __bf16* s = stage + lrow * NHID + kbA;
            a1[0].q[0] = *(const uint4*)(s);
            a1[0].q[1] = *(const uint4*)(s + 16);
            a1[1].q[0] = *(const uint4*)(s + 32);
            a1[1].q[1] = *(const uint4*)(s + 48);
        }

        // ---- layer 1 (WMMA) fused with layer 2 (64 -> 1 dot) ----
        float partial[8];
        #pragma unroll
        for (int i = 0; i < 8; ++i) partial[i] = 0.0f;

        #pragma unroll
        for (int nt = 0; nt < 4; ++nt) {
            f32x8 acc = {};
            #pragma unroll
            for (int t = 0; t < 2; ++t)
                acc = __builtin_amdgcn_wmma_f32_16x16x32_bf16(false, a1[t].v, false, bw1[t][nt].v,
                                                              (short)0, acc, false, false);
            #pragma unroll
            for (int i = 0; i < 8; ++i) {
                float hv = fmaxf(acc[i] + bias1[nt], 0.0f);
                partial[i] += hv * w2v[nt];                  // row-wise dot with w2
            }
        }

        // reduce each row over the 16 lanes of its half (xor within 16)
        #pragma unroll
        for (int i = 0; i < 8; ++i) {
            float v = partial[i];
            v += __shfl_xor(v, 1, 32);
            v += __shfl_xor(v, 2, 32);
            v += __shfl_xor(v, 4, 32);
            v += __shfl_xor(v, 8, 32);
            partial[i] = v;
        }
        if (lrow == 0) {
            #pragma unroll
            for (int i = 0; i < 8; ++i) {
                int row = tile * 16 + i + hi * 8;
                atomicAdd(&out[row], gate * (partial[i] + b2));
            }
        }
    }
}

extern "C" void kernel_launch(void* const* d_in, const int* in_sizes, int n_in,
                              void* d_out, int out_size, void* d_ws, size_t ws_size,
                              hipStream_t stream) {
    const int*   mains      = (const int*)  d_in[0];
    const int*   pairs      = (const int*)  d_in[1];
    const int*   pairs_list = (const int*)  d_in[2];
    const int*   offsets    = (const int*)  d_in[3];
    const float* embedding  = (const float*)d_in[4];
    const float* mw0 = (const float*)d_in[5];
    const float* mw1 = (const float*)d_in[6];
    const float* mw2 = (const float*)d_in[7];
    const float* mb0 = (const float*)d_in[8];
    const float* mb1 = (const float*)d_in[9];
    const float* mb2 = (const float*)d_in[10];
    const float* pw0 = (const float*)d_in[11];
    const float* pw1 = (const float*)d_in[12];
    const float* pw2 = (const float*)d_in[13];
    const float* pb0 = (const float*)d_in[14];
    const float* pb1 = (const float*)d_in[15];
    const float* pb2 = (const float*)d_in[16];
    const float* z_main  = (const float*)d_in[17];
    const float* z_pairs = (const float*)d_in[18];
    float* out = (float*)d_out;

    zero_f32<<<(out_size + 255) / 256, 256, 0, stream>>>(out, out_size);

    // main effects: 32 units, split batch 8x for occupancy
    dnamite_wmma<1><<<dim3(NFEAT, 8), 256, 0, stream>>>(
        mains, pairs_list, offsets, embedding,
        mw0, mw1, mw2, mb0, mb1, mb2, z_main, out);

    // pair effects: 496 units, split batch 2x
    dnamite_wmma<2><<<dim3(NPAIRS, 2), 256, 0, stream>>>(
        pairs, pairs_list, offsets, embedding,
        pw0, pw1, pw2, pb0, pb1, pb2, z_pairs, out);
}
